// SchNetForcePBC_7602092114193
// MI455X (gfx1250) — matrix operand order repaired and checked by reading, not compiled, not run on hardware
//
#include <hip/hip_runtime.h>
#include <hip/hip_bf16.h>

typedef __attribute__((ext_vector_type(16))) _Float16 v16h;
typedef __attribute__((ext_vector_type(8)))  _Float16 v8h;
typedef __attribute__((ext_vector_type(8)))  float    v8f;

#define N_ATOMS 800
#define FEAT    128
#define NG      50
#define CUTOFF_A 5.0f
#define GWIDTH  (5.0f/49.0f)
#define GGAMMA  (0.5f/(GWIDTH*GWIDTH))
#define LOG2F_  0.6931471805599453f

__device__ __forceinline__ float sspf(float v){
  // shifted softplus: log(1+exp(v)) - log(2)
  float sp = (v > 15.f) ? v : __logf(1.f + __expf(v));
  return sp - LOG2F_;
}

__device__ __forceinline__ v8f wmma16(v16h a, v16h b, v8f c){
  // D = A(16x32 f16) x B(32x16 f16) + C(16x16 f32)
  return __builtin_amdgcn_wmma_f32_16x16x32_f16(false, a, false, b, (short)0, c, false, false);
}

// ---------------------------------------------------------------- utilities
__global__ void zero_kernel(float* p, int n){
  int i = blockIdx.x*blockDim.x + threadIdx.x;
  if (i < n) p[i] = 0.f;
}

__global__ void embed_kernel(const int* __restrict__ Z, const float* __restrict__ emb,
                             float* __restrict__ x){
  int idx = blockIdx.x*blockDim.x + threadIdx.x;
  if (idx >= N_ATOMS*FEAT) return;
  int i = idx >> 7, f = idx & 127;
  x[idx] = emb[Z[i]*FEAT + f];
}

// Pre-swizzle a row-major weight W[K][128] into WMMA B-fragment-major layout:
// dst[((ks*8+nt)*32 + lane)*16 + t] = W[ks*32 + (t<8 ? t : t+8) + (lane<16 ? 0 : 8)][nt*16 + lane%16]
// (16-bit B 32x16 layout: lanes 0-15 hold K=0..7,16..23, lanes 16-31 hold K=8..15,24..31)
__global__ void swizzleB_kernel(const float* __restrict__ W, _Float16* __restrict__ dst,
                                int Ksrc, int Kpad){
  int idx = blockIdx.x*blockDim.x + threadIdx.x;
  int total = (Kpad/32)*8*32*16;
  if (idx >= total) return;
  int t    =  idx        & 15;
  int lane = (idx >> 4)  & 31;
  int nt   = (idx >> 9)  & 7;
  int ks   =  idx >> 12;
  int k = ks*32 + ((t < 8) ? t : t + 8) + ((lane < 16) ? 0 : 8);
  int n = nt*16 + (lane & 15);
  float v = (k < Ksrc) ? W[k*FEAT + n] : 0.f;
  dst[idx] = (_Float16)v;
}

// ------------------------------------------------- generic 128-wide WMMA GEMM
// C[M x 128] = post( A[M x 128](f32) @ Bfrag + bias )
// mode 0: plain store; mode 1: store ssp(.); mode 2: Cdst += result (residual)
__global__ __launch_bounds__(256) void gemm128_kernel(
    const float* __restrict__ A, const _Float16* __restrict__ Bfrag,
    const float* __restrict__ bias, float* __restrict__ Cdst,
    int Mtiles, int mode)
{
  int lane = threadIdx.x & 31;
  int wave = (blockIdx.x * blockDim.x + threadIdx.x) >> 5;
  if (wave >= Mtiles) return;
  int nl = lane & 15, hi = lane >> 4;
  int hioff = hi ? 8 : 0;
  const v16h* Bf = (const v16h*)Bfrag;
  int arow = wave*16 + nl;            // A row this lane feeds into its A-fragment

  v8f acc[8];
#pragma unroll
  for (int nt = 0; nt < 8; ++nt){
    float b = bias ? bias[nt*16 + nl] : 0.f;
#pragma unroll
    for (int v = 0; v < 8; ++v) acc[nt][v] = b;
  }

#pragma unroll
  for (int kc = 0; kc < 4; ++kc){
    int c0 = kc*32 + hioff;
    const float4* p0 = (const float4*)(A + arow*FEAT + c0);
    const float4* p1 = (const float4*)(A + arow*FEAT + c0 + 16);
    float4 q0 = p0[0], q1 = p0[1], q2 = p1[0], q3 = p1[1];
    float t0[8], t1[8];
    ((float4*)t0)[0] = q0; ((float4*)t0)[1] = q1;
    ((float4*)t1)[0] = q2; ((float4*)t1)[1] = q3;
    v16h a;
#pragma unroll
    for (int t = 0; t < 8; ++t){ a[t] = (_Float16)t0[t]; a[t+8] = (_Float16)t1[t]; }
#pragma unroll
    for (int nt = 0; nt < 8; ++nt){
      v16h b = Bf[(kc*8 + nt)*32 + lane];
      acc[nt] = wmma16(a, b, acc[nt]);
    }
  }

#pragma unroll
  for (int nt = 0; nt < 8; ++nt){
#pragma unroll
    for (int v = 0; v < 8; ++v){
      int r = wave*16 + v + hioff;
      int o = r*FEAT + nt*16 + nl;
      float val = acc[nt][v];
      if (mode == 1) val = sspf(val);
      if (mode == 2) Cdst[o] += val; else Cdst[o] = val;
    }
  }
}

// ------------------------------------------------------- fused cfconv kernel
// 256 threads = 8 waves; each wave owns two i atoms (w and w+8) of a 16-atom
// i-tile. All weight B-fragments staged once per block into LDS; per j-tile:
// Gaussian RBF A-fragments built analytically in registers -> GEMM1 (LDS B) ->
// ssp -> wave-private LDS relayout -> GEMM2 (LDS B) -> row mask -> elementwise
// reduce against f16 xf tile in LDS.
__global__ __launch_bounds__(256) void cfconv_kernel(
    const float* __restrict__ pos, const float* __restrict__ boxv,
    const float* __restrict__ xf,
    const _Float16* __restrict__ Wf1frag, const _Float16* __restrict__ Wf2frag,
    const float* __restrict__ bf1, const float* __restrict__ bf2,
    float* __restrict__ mout)
{
  __shared__ _Float16 wf1s[2*8*32*16];     // 16 KB: Wf1 B-fragments
  __shared__ _Float16 wf2s[4*8*32*16];     // 32 KB: Wf2 B-fragments
  __shared__ _Float16 xfj[16*FEAT];        //  4 KB: xf tile (f16) for 16 j atoms
  __shared__ _Float16 hlds[8*16*32];       //  8 KB: per-wave 16x32 h relayout slab

  int tid  = threadIdx.x;
  int lane = tid & 31;
  int w    = tid >> 5;                 // wave id 0..7
  int nl   = lane & 15;
  int hi   = lane >> 4;
  int hioff = hi ? 8 : 0;

  // stage weight fragments to LDS once per block (48 KB cooperative copy)
  {
    const uint4* s1 = (const uint4*)Wf1frag;
    uint4* d1 = (uint4*)wf1s;
    for (int e = tid; e < (2*8*32*16)/8; e += 256) d1[e] = s1[e];
    const uint4* s2 = (const uint4*)Wf2frag;
    uint4* d2 = (uint4*)wf2s;
    for (int e = tid; e < (4*8*32*16)/8; e += 256) d2[e] = s2[e];
  }

  const v16h* BF1 = (const v16h*)wf1s;
  const v16h* BF2 = (const v16h*)wf2s;

  float Lx = boxv[0]*10.f, Ly = boxv[4]*10.f, Lz = boxv[8]*10.f;
  int ig0 = blockIdx.x*16 + w;
  int ig1 = ig0 + 8;
  float pix[2], piy[2], piz[2];
  pix[0] = pos[ig0*3+0]*10.f; piy[0] = pos[ig0*3+1]*10.f; piz[0] = pos[ig0*3+2]*10.f;
  pix[1] = pos[ig1*3+0]*10.f; piy[1] = pos[ig1*3+1]*10.f; piz[1] = pos[ig1*3+2]*10.f;

  float macc[2][8];
#pragma unroll
  for (int ii = 0; ii < 2; ++ii)
#pragma unroll
    for (int nt = 0; nt < 8; ++nt) macc[ii][nt] = 0.f;

  _Float16* hw = hlds + w*(16*32);
  int jt_begin = blockIdx.y*10, jt_end = jt_begin + 10;

  for (int jt = jt_begin; jt < jt_end; ++jt){
    // stage xf tile as f16 (2048 elems; 8 per thread)
    __syncthreads();
    {
      const float4* s = (const float4*)(xf + jt*16*FEAT) + tid*2;
      float4 q0 = s[0], q1 = s[1];
      v8h hh;
      hh[0]=(_Float16)q0.x; hh[1]=(_Float16)q0.y; hh[2]=(_Float16)q0.z; hh[3]=(_Float16)q0.w;
      hh[4]=(_Float16)q1.x; hh[5]=(_Float16)q1.y; hh[6]=(_Float16)q1.z; hh[7]=(_Float16)q1.w;
      *(v8h*)&xfj[tid*8] = hh;
    }
    __syncthreads();

    int jg = jt*16 + nl;
    float pjx = pos[jg*3+0]*10.f, pjy = pos[jg*3+1]*10.f, pjz = pos[jg*3+2]*10.f;

#pragma unroll
    for (int ii = 0; ii < 2; ++ii){
      int ig = ii ? ig1 : ig0;
      // pairwise minimum-image distance: this wave's i atom vs j = lane%16
      float dx = pix[ii] - pjx; dx -= Lx*rintf(dx/Lx);
      float dy = piy[ii] - pjy; dy -= Ly*rintf(dy/Ly);
      float dz = piz[ii] - pjz; dz -= Lz*rintf(dz/Lz);
      float r  = sqrtf(fmaxf(dx*dx + dy*dy + dz*dz, 1e-12f));
      bool msk = (r < CUTOFF_A) && (ig != jg);
      unsigned mb = (unsigned)__ballot(msk);

      // Gaussian RBF -> two A-fragments (K = 0..31, 32..63; k>=50 zero-padded)
      // 16-bit A 16x32 layout: halves 0..7 -> K=hioff+t ; 8..15 -> K=16+hioff+(t-8)
      v16h a0, a1;
#pragma unroll
      for (int t = 0; t < 16; ++t){
        int kk = (t < 8) ? (hioff + t) : (16 + hioff + (t - 8));
        float v0 = 0.f, v1 = 0.f;
        if (msk){
          float dr0 = r - (float)kk*GWIDTH;
          v0 = __expf(-GGAMMA*dr0*dr0);
          int k1 = 32 + kk;
          if (k1 < NG){ float dr1 = r - (float)k1*GWIDTH; v1 = __expf(-GGAMMA*dr1*dr1); }
        }
        a0[t] = (_Float16)v0; a1[t] = (_Float16)v1;
      }

      // Wij accumulators for 8 N-tiles, init with bf2
      v8f d[8];
#pragma unroll
      for (int nt = 0; nt < 8; ++nt){
        float b = bf2[nt*16 + nl];
#pragma unroll
        for (int v = 0; v < 8; ++v) d[nt][v] = b;
      }

#pragma unroll
      for (int kc = 0; kc < 4; ++kc){
        // GEMM1: produce 32 columns of h = ssp(g@Wf1 + bf1), stash in LDS f16
#pragma unroll
        for (int u = 0; u < 2; ++u){
          int th = kc*2 + u;
          float b1 = bf1[th*16 + nl];
          v8f c;
#pragma unroll
          for (int v = 0; v < 8; ++v) c[v] = b1;
          c = wmma16(a0, BF1[(0*8 + th)*32 + lane], c);
          c = wmma16(a1, BF1[(1*8 + th)*32 + lane], c);
#pragma unroll
          for (int v = 0; v < 8; ++v)
            hw[(v + hioff)*32 + u*16 + nl] = (_Float16)sspf(c[v]);
        }
        asm volatile("s_wait_dscnt 0" ::: "memory");   // wave-private LDS RAW fence

        // reload as 16-bit A-fragment (row = lane%16, K halves per A layout)
        v8h lo8 = *(const v8h*)&hw[nl*32 + hioff];
        v8h hi8 = *(const v8h*)&hw[nl*32 + 16 + hioff];
        v16h ha;
#pragma unroll
        for (int t = 0; t < 8; ++t){ ha[t] = lo8[t]; ha[t+8] = hi8[t]; }

        // GEMM2 partial: Wij += h_chunk @ Wf2_chunk (B from LDS)
#pragma unroll
        for (int nt = 0; nt < 8; ++nt)
          d[nt] = wmma16(ha, BF2[(kc*8 + nt)*32 + lane], d[nt]);
      }

      // row mask + elementwise reduce: m[i,f] += sum_j Wij[j,f]*xf[j,f]
#pragma unroll
      for (int nt = 0; nt < 8; ++nt){
        float part = 0.f;
#pragma unroll
        for (int v = 0; v < 8; ++v){
          int rowj = v + hioff;
          float wv = ((mb >> rowj) & 1u) ? d[nt][v] : 0.f;
          part += wv * (float)xfj[rowj*FEAT + nt*16 + nl];
        }
        part += __shfl_xor(part, 16, 32);   // combine j rows 0-7 with 8-15
        macc[ii][nt] += part;
      }
    }
  }

  if (hi == 0){
#pragma unroll
    for (int nt = 0; nt < 8; ++nt){
      atomicAdd(&mout[ig0*FEAT + nt*16 + nl], macc[0][nt]);
      atomicAdd(&mout[ig1*FEAT + nt*16 + nl], macc[1][nt]);
    }
  }
}

// ------------------------------------------------------- output head (tiny)
__global__ void head_kernel(const float* __restrict__ x, const float* __restrict__ Wo1,
                            const float* __restrict__ bo1, const float* __restrict__ Wo2,
                            const float* __restrict__ bo2, float* __restrict__ out){
  __shared__ float red[64];
  int i = blockIdx.x;
  int k = threadIdx.x;   // 0..63
  float s = 0.f;
  for (int f = 0; f < FEAT; ++f) s += x[i*FEAT + f] * Wo1[f*64 + k];
  red[k] = sspf(s + bo1[k]) * Wo2[k];
  __syncthreads();
  if (k == 0){
    float e = bo2[0];
    for (int q = 0; q < 64; ++q) e += red[q];
    atomicAdd(out, e);
  }
}

// --------------------------------------------------------------------- host
extern "C" void kernel_launch(void* const* d_in, const int* in_sizes, int n_in,
                              void* d_out, int out_size, void* d_ws, size_t ws_size,
                              hipStream_t stream) {
  (void)in_sizes; (void)n_in; (void)out_size; (void)ws_size;
  const float* positions = (const float*)d_in[0];
  const float* boxv      = (const float*)d_in[1];
  const int*   Z         = (const int*)  d_in[2];
  const float* emb       = (const float*)d_in[3];
  const float* Wf1       = (const float*)d_in[4];
  const float* bf1       = (const float*)d_in[5];
  const float* Wf2       = (const float*)d_in[6];
  const float* bf2       = (const float*)d_in[7];
  const float* Win       = (const float*)d_in[8];
  const float* W2        = (const float*)d_in[9];
  const float* b2        = (const float*)d_in[10];
  const float* W3        = (const float*)d_in[11];
  const float* b3        = (const float*)d_in[12];
  const float* Wo1       = (const float*)d_in[13];
  const float* bo1       = (const float*)d_in[14];
  const float* Wo2       = (const float*)d_in[15];
  const float* bo2       = (const float*)d_in[16];
  float* out = (float*)d_out;

  char* ws = (char*)d_ws;
  size_t off = 0;
  float* x   = (float*)(ws + off); off += (size_t)N_ATOMS*FEAT*sizeof(float);
  float* xfb = (float*)(ws + off); off += (size_t)N_ATOMS*FEAT*sizeof(float);
  float* mb_ = (float*)(ws + off); off += (size_t)N_ATOMS*FEAT*sizeof(float);
  float* tb  = (float*)(ws + off); off += (size_t)N_ATOMS*FEAT*sizeof(float);
  _Float16* fr = (_Float16*)(ws + off);
  _Float16 *fWf1[3], *fWf2[3], *fWin[3], *fW2[3], *fW3[3];
  for (int l = 0; l < 3; ++l){
    fWf1[l] = fr; fr += 2*8*32*16;   // K=64
    fWin[l] = fr; fr += 4*8*32*16;   // K=128
    fWf2[l] = fr; fr += 4*8*32*16;
    fW2[l]  = fr; fr += 4*8*32*16;
    fW3[l]  = fr; fr += 4*8*32*16;
  }

  embed_kernel<<<(N_ATOMS*FEAT + 255)/256, 256, 0, stream>>>(Z, emb, x);
  for (int l = 0; l < 3; ++l){
    swizzleB_kernel<<<(8192 + 255)/256, 256, 0, stream>>>(Wf1 + l*NG*FEAT,  fWf1[l], NG,   64);
    swizzleB_kernel<<<(16384 + 255)/256, 256, 0, stream>>>(Win + l*FEAT*FEAT, fWin[l], FEAT, FEAT);
    swizzleB_kernel<<<(16384 + 255)/256, 256, 0, stream>>>(Wf2 + l*FEAT*FEAT, fWf2[l], FEAT, FEAT);
    swizzleB_kernel<<<(16384 + 255)/256, 256, 0, stream>>>(W2  + l*FEAT*FEAT, fW2[l],  FEAT, FEAT);
    swizzleB_kernel<<<(16384 + 255)/256, 256, 0, stream>>>(W3  + l*FEAT*FEAT, fW3[l],  FEAT, FEAT);
  }

  const int MT = N_ATOMS/16;     // 50 M-tiles
  for (int l = 0; l < 3; ++l){
    gemm128_kernel<<<7, 256, 0, stream>>>(x, fWin[l], nullptr, xfb, MT, 0);       // xf = x@Win
    zero_kernel<<<(N_ATOMS*FEAT + 255)/256, 256, 0, stream>>>(mb_, N_ATOMS*FEAT);
    cfconv_kernel<<<dim3(MT, 5), 256, 0, stream>>>(positions, boxv, xfb,
                                                   fWf1[l], fWf2[l],
                                                   bf1 + l*FEAT, bf2 + l*FEAT, mb_);
    gemm128_kernel<<<7, 256, 0, stream>>>(mb_, fW2[l], b2 + l*FEAT, tb, MT, 1);   // t = ssp(m@W2+b2)
    gemm128_kernel<<<7, 256, 0, stream>>>(tb,  fW3[l], b3 + l*FEAT, x,  MT, 2);   // x += t@W3+b3
  }

  zero_kernel<<<1, 32, 0, stream>>>(out, 1);
  head_kernel<<<N_ATOMS, 64, 0, stream>>>(x, Wo1, bo1, Wo2, bo2, out);
}
